// MultiHeadAttention_9131100471609
// MI455X (gfx1250) — matrix operand order repaired and checked by reading, not compiled, not run on hardware
//
#include <hip/hip_runtime.h>
#include <hip/hip_bf16.h>
#include <cstdint>
#include <cstddef>

// Problem constants (from reference): S=2048, B=4, F=512, H=8, D=64
#define S_LEN 2048
#define BATCH 4
#define FDIM  512
#define NHEAD 8
#define HDIM  64
#define BHEAD (BATCH * NHEAD)   // 32
#define MROWS (S_LEN * BATCH)   // 8192

typedef __attribute__((ext_vector_type(16))) __bf16 bf16x16;
typedef __attribute__((ext_vector_type(8)))  float  f32x8;
typedef __attribute__((ext_vector_type(4)))  unsigned int u32x4;
typedef __attribute__((ext_vector_type(8)))  unsigned int u32x8;

union AFrag { bf16x16 v; uint4 q[2]; __bf16 e[16]; };
union CFrag { f32x8 v; float f[8]; };

// B-matrix fragment per lane: 8 bf16 at p, 8 bf16 at p+16 (same K split as A)
__device__ __forceinline__ bf16x16 ld16(const __bf16* p) {
  AFrag u;
  u.q[0] = *(const uint4*)p;
  u.q[1] = *(const uint4*)(p + 16);
  return u.v;
}
// A-matrix fragment per lane: 8 bf16 at p, 8 bf16 at p+16 (K split per ISA layout)
__device__ __forceinline__ bf16x16 ld8_8(const __bf16* p) {
  AFrag u;
  u.q[0] = *(const uint4*)p;
  u.q[1] = *(const uint4*)(p + 16);
  return u.v;
}

__device__ __forceinline__ f32x8 wmma_bf16(bf16x16 a, bf16x16 b, f32x8 c) {
  // (neg_a, A, neg_b, B, c_mod, C, reuse_a, reuse_b)
  return __builtin_amdgcn_wmma_f32_16x16x32_bf16(false, a, false, b, (short)0, c,
                                                 false, false);
}

// CDNA5 async global->LDS copy (ASYNCcnt-tracked, bypasses VGPRs).
// LDS generic pointers: addr[31:0] is the LDS byte offset (ISA 10.2 aperture rule).
__device__ __forceinline__ void async_copy_b128(const void* gptr, void* lptr) {
  unsigned      lds = (unsigned)(uintptr_t)lptr;
  unsigned long ga  = (unsigned long)(uintptr_t)gptr;
  asm volatile("global_load_async_to_lds_b128 %0, %1, off"
               :: "v"(lds), "v"(ga) : "memory");
}
__device__ __forceinline__ void wait_async0() {
  asm volatile("s_wait_asynccnt 0" ::: "memory");
}

// CDNA5 Tensor Data Mover: 2D tile load Global->LDS with hardware LDS row
// padding. Two-group D# (tensors up to 2D), workgroup_mask=0 (not in cluster).
// data_size = 2 bytes (bf16). Tracked with TENSORcnt.
__device__ __forceinline__ void tdm_load_2d_bf16(unsigned lds_off, const void* gptr,
                                                 unsigned tensor_d0, unsigned tensor_d1,
                                                 unsigned tile_d0, unsigned tile_d1,
                                                 unsigned stride_d0,
                                                 unsigned pad_interval_code,
                                                 unsigned pad_amount_code) {
  unsigned long ga = (unsigned long)(uintptr_t)gptr;
  u32x4 g0;
  g0[0] = 1u;                                    // count = 1 valid descriptor
  g0[1] = lds_off;                               // lds_addr (bytes)
  g0[2] = (unsigned)ga;                          // global_addr[31:0]
  g0[3] = ((unsigned)(ga >> 32) & 0x01FFFFFFu)   // global_addr[56:32]
        | (2u << 30);                            // type = 2 ("image")
  u32x8 g1;
  g1[0] = (1u << 16)                             // data_size = 1 -> 2 bytes
        | (pad_amount_code ? (1u << 20) : 0u)    // pad_enable
        | (pad_interval_code << 22)              // DWORDs before pad: code 4 = 32
        | (pad_amount_code << 25);               // pad DWORDs: code 3 = 4
  g1[1] = (tensor_d0 & 0xFFFFu) << 16;           // tensor_dim0[15:0]  @ [63:48]
  g1[2] = ((tensor_d0 >> 16) & 0xFFFFu)          // tensor_dim0[31:16] @ [79:64]
        | ((tensor_d1 & 0xFFFFu) << 16);         // tensor_dim1[15:0]  @ [95:80]
  g1[3] = ((tensor_d1 >> 16) & 0xFFFFu)          // tensor_dim1[31:16] @ [111:96]
        | (tile_d0 << 16);                       // tile_dim0          @ [127:112]
  g1[4] = tile_d1;                               // tile_dim1 @ [143:128], tile_dim2=0
  g1[5] = stride_d0;                             // tensor_dim0_stride[31:0]
  g1[6] = 0u;                                    // stride hi / dim1_stride lo
  g1[7] = 0u;                                    // dim1_stride hi
  asm volatile("tensor_load_to_lds %0, %1" :: "s"(g0), "s"(g1) : "memory");
}

// wave32: reduce within each 16-lane half (the lanes sharing a C-matrix row)
__device__ __forceinline__ float redmax16(float v) {
#pragma unroll
  for (int o = 1; o < 16; o <<= 1) v = fmaxf(v, __shfl_xor(v, o, 32));
  return v;
}
__device__ __forceinline__ float redsum16(float v) {
#pragma unroll
  for (int o = 1; o < 16; o <<= 1) v += __shfl_xor(v, o, 32);
  return v;
}

// -----------------------------------------------------------------------------
// GEMM: out[m, g] = relu( sum_f X[m, f] * W[g, f] + bias[g] )
//   M = 8192 (= S*B rows, row-major over f), N = K = 512.
// Block tile 128(M) x 64(N), K-step 64 (8 WMMAs per barrier pair, 8 iters).
// 256 threads = 8 waves, wave tile 32x32.
// EPI == 0: write bf16 into attention layout (B*H, S, D), optional RoPE.
// EPI == 1: write fp32 row-major (final output); X is bf16 -> async staging.
// -----------------------------------------------------------------------------
template <typename TX, int EPI>
__global__ __launch_bounds__(256) void gemm_proj(const TX* __restrict__ X,
                                                 const float* __restrict__ W,
                                                 const float* __restrict__ bias,
                                                 void* __restrict__ outp,
                                                 int do_rope) {
  __shared__ __bf16 Xs[128][72];  // 144B row stride -> 16B aligned b128 DS ops
  __shared__ __bf16 Ws[64][72];

  const int tid  = threadIdx.x;
  const int lane = tid & 31;
  const int w    = tid >> 5;
  const int lm   = lane & 15;     // N-column / M-row selector within fragment
  const int lh   = lane >> 4;     // which K/M half
  const int wm   = (w >> 1) * 32; // wave M offset in block tile
  const int wn   = (w & 1) * 32;  // wave N offset in block tile
  const int m0   = blockIdx.y * 128;
  const int n0   = blockIdx.x * 64;

  CFrag acc[2][2];
#pragma unroll
  for (int a = 0; a < 2; a++)
#pragma unroll
    for (int b = 0; b < 2; b++)
#pragma unroll
      for (int r = 0; r < 8; r++) acc[a][b].f[r] = 0.0f;

  for (int k0 = 0; k0 < FDIM; k0 += 64) {
    // ---- stage X tile (128 x 64) into LDS as bf16 ----
    if constexpr (sizeof(TX) == 4) {
#pragma unroll
      for (int i = 0; i < 8; i++) {
        int slot = tid + 256 * i;           // 2048 float4 slots (16 per row)
        int row = slot >> 4, col = (slot & 15) * 4;
        float4 fv = *(const float4*)((const float*)X +
                                     (size_t)(m0 + row) * FDIM + k0 + col);
        Xs[row][col + 0] = (__bf16)fv.x;
        Xs[row][col + 1] = (__bf16)fv.y;
        Xs[row][col + 2] = (__bf16)fv.z;
        Xs[row][col + 3] = (__bf16)fv.w;
      }
    } else {
      // pure bf16 copy: CDNA5 async global->LDS (no VGPR round-trip)
#pragma unroll
      for (int i = 0; i < 4; i++) {
        int slot = tid + 256 * i;           // 1024 b128 slots (8 per row)
        int row = slot >> 3, col = (slot & 7) * 8;
        async_copy_b128((const __bf16*)X + (size_t)(m0 + row) * FDIM + k0 + col,
                        &Xs[row][col]);
      }
    }
    // ---- stage W tile (64 x 64): rows are output features g ----
#pragma unroll
    for (int i = 0; i < 4; i++) {
      int slot = tid + 256 * i;             // 1024 float4 slots (16 per row)
      int row = slot >> 4, col = (slot & 15) * 4;
      float4 fv = *(const float4*)(W + (size_t)(n0 + row) * FDIM + k0 + col);
      Ws[row][col + 0] = (__bf16)fv.x;
      Ws[row][col + 1] = (__bf16)fv.y;
      Ws[row][col + 2] = (__bf16)fv.z;
      Ws[row][col + 3] = (__bf16)fv.w;
    }
    if constexpr (sizeof(TX) == 2) wait_async0();
    __syncthreads();

#pragma unroll
    for (int kc = 0; kc < 2; kc++) {
      // A: 16x32 per tile (row = M per lane, K split 8+8 across lane halves)
      bf16x16 a0 = ld8_8(&Xs[wm + lm][kc * 32 + lh * 8]);
      bf16x16 a1 = ld8_8(&Xs[wm + 16 + lm][kc * 32 + lh * 8]);
      // B: 32x16 per tile (lane = N column, K split 8+8 across lane halves)
      bf16x16 b0 = ld16(&Ws[wn + lm][kc * 32 + lh * 8]);
      bf16x16 b1 = ld16(&Ws[wn + 16 + lm][kc * 32 + lh * 8]);
      acc[0][0].v = wmma_bf16(a0, b0, acc[0][0].v);
      acc[0][1].v = wmma_bf16(a0, b1, acc[0][1].v);
      acc[1][0].v = wmma_bf16(a1, b0, acc[1][0].v);
      acc[1][1].v = wmma_bf16(a1, b1, acc[1][1].v);
    }
    __syncthreads();
  }

  // ---- epilogue: bias + relu (+ RoPE) + store ----
#pragma unroll
  for (int at = 0; at < 2; at++)
#pragma unroll
    for (int bt = 0; bt < 2; bt++)
#pragma unroll
      for (int r = 0; r < 8; r++) {
        int mg = m0 + wm + at * 16 + r + 8 * lh;  // global row = s*B + b
        int ng = n0 + wn + bt * 16 + lm;          // global feature g = h*D + d
        float v = fmaxf(acc[at][bt].f[r] + bias[ng], 0.0f);
        if constexpr (EPI == 0) {
          int s = mg >> 2, b = mg & 3;
          int h = ng >> 6, d = ng & 63;
          if (do_rope) {
            // even/odd feature pair lives in adjacent lanes -> shfl_xor(1)
            float partner = __shfl_xor(v, 1, 32);
            int p = d >> 1;
            // theta_p = 10000^(-2p/64) = exp2(-p * log2(10000) / 32)
            float theta = exp2f((float)p * (-13.287712379549449f / 32.0f));
            float ang = (float)s * theta;
            float cs = __cosf(ang), sn = __sinf(ang);
            v = ((d & 1) == 0) ? (v * cs - partner * sn)   // even: r*c - i*s
                               : (partner * sn + v * cs);  // odd:  r*s + i*c
          }
          ((__bf16*)outp)[(((size_t)(b * NHEAD + h)) * S_LEN + s) * HDIM + d] =
              (__bf16)v;
        } else {
          ((float*)outp)[(size_t)mg * FDIM + ng] = v;
        }
      }
}

// -----------------------------------------------------------------------------
// Flash attention: grid (S/128, B*H). 8 waves/block, each wave owns 16 queries.
// K tile staged by the Tensor Data Mover (one tensor_load_to_lds per block per
// tile, hardware applies the 144B LDS row padding); V staged via b128 reg loads
// + transpose scatter. Online softmax with wave-half shuffles; P routed through
// wave-private LDS (same-wave DS ordering -> no barrier) to become A-fragments.
// -----------------------------------------------------------------------------
__global__ __launch_bounds__(256) void attn_kernel(const __bf16* __restrict__ Qr,
                                                   const __bf16* __restrict__ Kr,
                                                   const __bf16* __restrict__ Vr,
                                                   __bf16* __restrict__ Y) {
  __shared__ __bf16 Ks[64][72];      // [key][d]  (TDM fills, incl. padding)
  __shared__ __bf16 Vt[64][72];      // [d][key]  (transposed at load)
  __shared__ __bf16 Pw[8][16][72];   // per-wave P scratch [qrow][key]

  const int tid  = threadIdx.x;
  const int lane = tid & 31;
  const int w    = tid >> 5;
  const int lm   = lane & 15;
  const int lh   = lane >> 4;
  const int bh   = blockIdx.y;
  const int qw0  = blockIdx.x * 128 + w * 16;

  // Q fragments (A-layout) held in registers for the whole pass
  const __bf16* qbase = Qr + ((size_t)bh * S_LEN + qw0 + lm) * HDIM;
  bf16x16 qa0 = ld8_8(qbase + lh * 8);        // d = 0..31
  bf16x16 qa1 = ld8_8(qbase + 32 + lh * 8);   // d = 32..63

  CFrag O[4];
  float mrun[8], lrun[8];
#pragma unroll
  for (int dn = 0; dn < 4; dn++)
#pragma unroll
    for (int r = 0; r < 8; r++) O[dn].f[r] = 0.0f;
#pragma unroll
  for (int r = 0; r < 8; r++) { mrun[r] = -1e30f; lrun[r] = 0.0f; }

  const int ldrow = tid >> 2;           // 0..63 key row
  const int ldseg = (tid & 3) * 16;     // 16-element segment of d

  for (int kt = 0; kt < S_LEN; kt += 64) {
    // ---- K tile: Tensor Data Mover, one descriptor per block (wave 0) ----
    if (w == 0) {
      tdm_load_2d_bf16((unsigned)(uintptr_t)&Ks[0][0],
                       Kr + ((size_t)bh * S_LEN + kt) * HDIM,
                       /*tensor_d0=*/HDIM, /*tensor_d1=*/BHEAD * S_LEN,
                       /*tile_d0=*/HDIM, /*tile_d1=*/64,
                       /*stride_d0=*/HDIM,
                       /*pad_interval(32 DW)=*/4u, /*pad_amount(4 DW)=*/3u);
    }
    // ---- V tile: b128 reg loads, transpose into LDS ----
    const __bf16* vsrc = Vr + ((size_t)bh * S_LEN + kt + ldrow) * HDIM + ldseg;
    AFrag vf;
    vf.q[0] = *(const uint4*)vsrc;
    vf.q[1] = *(const uint4*)(vsrc + 8);
#pragma unroll
    for (int j = 0; j < 16; j++) Vt[ldseg + j][ldrow] = vf.e[j];
    if (kt + 64 < S_LEN) {  // global_prefetch_b8 of next V tile
      __builtin_prefetch(vsrc + 64 * HDIM, 0, 1);
    }
    if (w == 0) __builtin_amdgcn_s_wait_tensorcnt(0);
    __syncthreads();

    // ---- scores: S = Q (16x64) . K^T (64x64), 4 sub-tiles of 16 keys ----
    CFrag Sc[4];
#pragma unroll
    for (int nc = 0; nc < 4; nc++) {
      CFrag c;
#pragma unroll
      for (int r = 0; r < 8; r++) c.f[r] = 0.0f;
      bf16x16 kb0 = ld16(&Ks[nc * 16 + lm][lh * 8]);       // d 0..31
      bf16x16 kb1 = ld16(&Ks[nc * 16 + lm][32 + lh * 8]);  // d 32..63
      c.v = wmma_bf16(qa0, kb0, c.v);
      c.v = wmma_bf16(qa1, kb1, c.v);
#pragma unroll
      for (int r = 0; r < 8; r++) Sc[nc].f[r] = c.f[r] * 0.125f;  // 1/sqrt(64)
    }

    // ---- online softmax update ----
    float scalef[8];
#pragma unroll
    for (int r = 0; r < 8; r++) {
      float mx = fmaxf(fmaxf(Sc[0].f[r], Sc[1].f[r]),
                       fmaxf(Sc[2].f[r], Sc[3].f[r]));
      mx = redmax16(mx);
      float nm = fmaxf(mrun[r], mx);
      scalef[r] = __expf(mrun[r] - nm);
      mrun[r] = nm;
    }
    float psum[8] = {0.f, 0.f, 0.f, 0.f, 0.f, 0.f, 0.f, 0.f};
#pragma unroll
    for (int nc = 0; nc < 4; nc++)
#pragma unroll
      for (int r = 0; r < 8; r++) {
        float p = __expf(Sc[nc].f[r] - mrun[r]);
        psum[r] += p;
        // C-layout (row = r + 8*lh, col = lane) -> LDS -> A-layout reload
        Pw[w][r + 8 * lh][nc * 16 + lm] = (__bf16)p;
      }
#pragma unroll
    for (int r = 0; r < 8; r++) {
      lrun[r] = lrun[r] * scalef[r] + redsum16(psum[r]);
#pragma unroll
      for (int dn = 0; dn < 4; dn++) O[dn].f[r] *= scalef[r];
    }

    // ---- O += P (16x64) . V (64x64); same-wave DS ordering, no barrier ----
    bf16x16 pa0 = ld8_8(&Pw[w][lm][lh * 8]);        // keys 0..31
    bf16x16 pa1 = ld8_8(&Pw[w][lm][32 + lh * 8]);   // keys 32..63
#pragma unroll
    for (int dn = 0; dn < 4; dn++) {
      bf16x16 vb0 = ld16(&Vt[dn * 16 + lm][lh * 8]);
      bf16x16 vb1 = ld16(&Vt[dn * 16 + lm][32 + lh * 8]);
      O[dn].v = wmma_bf16(pa0, vb0, O[dn].v);
      O[dn].v = wmma_bf16(pa1, vb1, O[dn].v);
    }
    __syncthreads();
  }

  // ---- normalize and write Y in (S*B, F) bf16 row-major ----
  const int b = bh >> 3, h = bh & 7;
#pragma unroll
  for (int dn = 0; dn < 4; dn++)
#pragma unroll
    for (int r = 0; r < 8; r++) {
      float v = O[dn].f[r] / lrun[r];
      int s = qw0 + r + 8 * lh;
      int d = dn * 16 + lm;
      Y[((size_t)(s * BATCH + b)) * FDIM + h * HDIM + d] = (__bf16)v;
    }
}

// -----------------------------------------------------------------------------
extern "C" void kernel_launch(void* const* d_in, const int* in_sizes, int n_in,
                              void* d_out, int out_size, void* d_ws, size_t ws_size,
                              hipStream_t stream) {
  (void)in_sizes; (void)n_in; (void)out_size; (void)ws_size;
  const float* q  = (const float*)d_in[0];
  const float* k  = (const float*)d_in[1];
  const float* v  = (const float*)d_in[2];
  const float* Wq = (const float*)d_in[3];
  const float* bq = (const float*)d_in[4];
  const float* Wk = (const float*)d_in[5];
  const float* bk = (const float*)d_in[6];
  const float* Wv = (const float*)d_in[7];
  const float* bv = (const float*)d_in[8];
  const float* Wo = (const float*)d_in[9];
  const float* bo = (const float*)d_in[10];

  const size_t nQKV = (size_t)BHEAD * S_LEN * HDIM;  // 4M elems (8 MB bf16)
  __bf16* Qr = (__bf16*)d_ws;
  __bf16* Kr = Qr + nQKV;
  __bf16* Vr = Kr + nQKV;
  __bf16* Yb = Vr + nQKV;  // (S*B, F) bf16, 8 MB  -> 32 MB workspace total

  dim3 blk(256);
  dim3 gproj(FDIM / 64, MROWS / 128);  // (8, 64)
  gemm_proj<float, 0><<<gproj, blk, 0, stream>>>(q, Wq, bq, (void*)Qr, 1);
  gemm_proj<float, 0><<<gproj, blk, 0, stream>>>(k, Wk, bk, (void*)Kr, 1);
  gemm_proj<float, 0><<<gproj, blk, 0, stream>>>(v, Wv, bv, (void*)Vr, 0);
  attn_kernel<<<dim3(S_LEN / 128, BHEAD), blk, 0, stream>>>(Qr, Kr, Vr, Yb);
  gemm_proj<__bf16, 1><<<gproj, blk, 0, stream>>>(Yb, Wo, bo, d_out, 0);
}
